// AGREE_41205916237970
// MI455X (gfx1250) — compile-verified, matrix operand order checked by reading, not run on hardware
//
#include <hip/hip_runtime.h>
#include <hip/hip_bf16.h>

typedef __attribute__((ext_vector_type(16))) _Float16 v16h;
typedef __attribute__((ext_vector_type(8)))  _Float16 v8h;
typedef __attribute__((ext_vector_type(8)))  float    v8f;

#define B_TOT 65536
#define GPB   32            // groups per block (8 waves x 4 groups/tile)
#define ROWS  (GPB * 4)     // 128 member-rows per block
#define NTHR  256

// flat shared pointer -> 32-bit LDS address (addrspace(3) value lives in low bits)
#define LDS_ADDR(p) ((unsigned)(uintptr_t)(p))

__global__ __launch_bounds__(NTHR) void agree_fused(
    const int*   __restrict__ group_inputs,
    const int*   __restrict__ item_inputs,
    const int*   __restrict__ group_members,
    const float* __restrict__ user_emb,
    const float* __restrict__ item_emb,
    const float* __restrict__ group_emb,
    const float* __restrict__ att_w1,
    const float* __restrict__ att_b1,
    const float* __restrict__ att_w2,
    const float* __restrict__ att_b2,
    const float* __restrict__ cls_w,
    const float* __restrict__ cls_b,
    const float* __restrict__ pred_w1,
    const float* __restrict__ pred_b1,
    const float* __restrict__ pred_w2,
    const float* __restrict__ pred_b2,
    float*       __restrict__ out)
{
  // att_w1 transposed to n-major f16 so B-fragments are contiguous 16B LDS loads
  __shared__ __align__(16) _Float16 sh_w1t[16 * 512];   // 16 KB
  __shared__ __align__(16) float    sh_pw1[768 * 8];    // 24 KB (filled by async DMA)
  __shared__ float sh_b1[16], sh_w2[16];
  __shared__ float sh_pb1[8], sh_pw2[8];
  __shared__ float sh_misc[8];        // 0:att_b2 1:cw0 2:cw1 3:cb0 4:cb1 5:pred_b2
  __shared__ float sh_logits[ROWS];
  __shared__ float sh_atwt[GPB * 4];
  __shared__ int   sh_meta[GPB];
  __shared__ int   sh_uid[ROWS];
  __shared__ int   sh_iid[GPB];
  __shared__ int   sh_gid[GPB];

  const int tid  = threadIdx.x;
  const int wave = tid >> 5;
  const int lane = tid & 31;
  const int bg0  = blockIdx.x * GPB;

  // ---- phase 0a: async DMA pred_w1 (24 KB contiguous) straight into LDS ----
  {
    const unsigned ldst = LDS_ADDR(sh_pw1) + (unsigned)tid * 16u;
    const unsigned goff = (unsigned)tid * 16u;
    #pragma unroll
    for (int it = 0; it < 6; ++it) {
      unsigned l = ldst + (unsigned)it * 4096u;
      unsigned g = goff + (unsigned)it * 4096u;
      asm volatile("global_load_async_to_lds_b128 %0, %1, %2"
                   :: "v"(l), "v"(g), "s"(pred_w1)
                   : "memory");
    }
  }

  // ---- phase 0b: stage remaining weights + indices (overlaps with async DMA) ----
  for (int i = tid; i < 16 * 512; i += NTHR) {
    int n = i >> 9, k = i & 511;
    sh_w1t[i] = (_Float16)att_w1[k * 16 + n];
  }
  if (tid < 16) { sh_b1[tid] = att_b1[tid]; sh_w2[tid] = att_w2[tid]; }
  else if (tid >= 32 && tid < 40) { sh_pb1[tid - 32] = pred_b1[tid - 32]; sh_pw2[tid - 32] = pred_w2[tid - 32]; }
  else if (tid == 64) {
    sh_misc[0] = att_b2[0];
    sh_misc[1] = cls_w[0]; sh_misc[2] = cls_w[1];
    sh_misc[3] = cls_b[0]; sh_misc[4] = cls_b[1];
    sh_misc[5] = pred_b2[0];
  }
  if (tid < GPB) {
    int bglob = bg0 + tid;
    int gid = group_inputs[bglob];
    sh_gid[tid] = gid;
    sh_iid[tid] = item_inputs[bglob];
    // group_emb row is only consumed in phase 3 -> prefetch now
    __builtin_prefetch(group_emb + (size_t)gid * 256, 0, 1);
    #pragma unroll
    for (int s = 0; s < 4; s++) sh_uid[tid * 4 + s] = group_members[gid * 4 + s];
  }
  // drain the async LDS DMA before anyone reads sh_pw1
  asm volatile("s_wait_asynccnt 0" ::: "memory");
  __syncthreads();

  // ---- phase 1: attention GEMM per wave: [16 rows x K=512] @ [512 x 16] via WMMA ----
  {
    const int halfsel = lane >> 4;       // K-half selector (A/B fragment layout)
    const int rloc    = lane & 15;       // tile row for A, feature col for B
    const int rowA    = wave * 16 + rloc;
    const float* mrow = user_emb + (size_t)sh_uid[rowA] * 256;
    const float* irow = item_emb + (size_t)sh_iid[rowA >> 2] * 256;
    const _Float16* wn = sh_w1t + (size_t)rloc * 512;

    v8f acc = {};
    #pragma unroll
    for (int kk = 0; kk < 16; kk++) {
      const int k0 = kk * 32 + halfsel * 8;   // elements 0..7 of fragment
      const int k1 = k0 + 16;                 // elements 8..15 of fragment
      // rows of gi = concat(member[256], item[256])
      const float* src = (kk < 8) ? mrow : (irow - 256);
      v8f alo = *(const v8f*)(src + k0);
      v8f ahi = *(const v8f*)(src + k1);
      v16h a;
      #pragma unroll
      for (int i = 0; i < 8; i++) { a[i] = (_Float16)alo[i]; a[8 + i] = (_Float16)ahi[i]; }
      v8h blo = *(const v8h*)(wn + k0);
      v8h bhi = *(const v8h*)(wn + k1);
      v16h bm;
      #pragma unroll
      for (int i = 0; i < 8; i++) { bm[i] = blo[i]; bm[8 + i] = bhi[i]; }
      acc = __builtin_amdgcn_wmma_f32_16x16x32_f16(
          /*neg_a=*/false, a, /*neg_b=*/false, bm,
          /*c_mod=*/(short)0, acc, /*reuse_a=*/false, /*reuse_b=*/false);
    }

    // epilogue: h = relu(acc + b1), logit = h . att_w2 + b2
    // C layout: vgpr i -> row (i + 8*halfsel), feature = lane&15
    const float b1n = sh_b1[rloc], w2n = sh_w2[rloc], ab2 = sh_misc[0];
    #pragma unroll
    for (int i = 0; i < 8; i++) {
      float t = fmaxf(acc[i] + b1n, 0.0f) * w2n;
      t += __shfl_xor(t, 1, 32);
      t += __shfl_xor(t, 2, 32);
      t += __shfl_xor(t, 4, 32);
      t += __shfl_xor(t, 8, 32);
      if (rloc == 0) sh_logits[wave * 16 + halfsel * 8 + i] = t + ab2;
    }
  }
  __syncthreads();

  // ---- phase 2: softmax / argmax / classifier per group ----
  if (tid < GPB) {
    float l[4];
    #pragma unroll
    for (int s = 0; s < 4; s++) l[s] = sh_logits[tid * 4 + s];
    int idx = 0; float mx = l[0];
    #pragma unroll
    for (int s = 1; s < 4; s++) if (l[s] > mx) { mx = l[s]; idx = s; }  // first-max
    float e[4], sum = 0.0f;
    #pragma unroll
    for (int s = 0; s < 4; s++) { e[s] = expf(l[s] - mx); sum += e[s]; }
    float inv = 1.0f / sum;
    int bglob = bg0 + tid;
    #pragma unroll
    for (int s = 0; s < 4; s++) {
      float aw = e[s] * inv;
      sh_atwt[tid * 4 + s] = aw;
      out[B_TOT + bglob * 4 + s] = aw;            // at_wt section
    }
    float atm = e[idx] * inv;
    float s0 = atm * sh_misc[1] + sh_misc[3];
    float s1 = atm * sh_misc[2] + sh_misc[4];
    int cls = (s1 > s0) ? 1 : 0;                  // argmax over 2, first occurrence
    out[5 * B_TOT + bglob] = (float)cls;          // pred_class section
    sh_meta[tid] = idx | (cls << 2);
  }
  __syncthreads();

  // ---- phase 3: group vector + prediction MLP (each wave handles 4 groups) ----
  const float pb2 = sh_misc[5];
  for (int q = 0; q < GPB / 8; q++) {
    const int g = wave * (GPB / 8) + q;
    const int bglob = bg0 + g;
    const int meta = sh_meta[g];
    const int midx = meta & 3;
    const int isleader = meta >> 2;
    // branchless leader-vs-weighted: one-hot weights when leader
    float w0 = isleader ? (midx == 0 ? 1.0f : 0.0f) : sh_atwt[g * 4 + 0];
    float w1 = isleader ? (midx == 1 ? 1.0f : 0.0f) : sh_atwt[g * 4 + 1];
    float w2 = isleader ? (midx == 2 ? 1.0f : 0.0f) : sh_atwt[g * 4 + 2];
    float w3 = isleader ? (midx == 3 ? 1.0f : 0.0f) : sh_atwt[g * 4 + 3];
    const float* m0 = user_emb + (size_t)sh_uid[g * 4 + 0] * 256;
    const float* m1 = user_emb + (size_t)sh_uid[g * 4 + 1] * 256;
    const float* m2 = user_emb + (size_t)sh_uid[g * 4 + 2] * 256;
    const float* m3 = user_emb + (size_t)sh_uid[g * 4 + 3] * 256;
    const float* grow = group_emb + (size_t)sh_gid[g] * 256;
    const float* irow = item_emb + (size_t)sh_iid[g] * 256;

    const int d0 = lane * 8;   // 8 contiguous dims per lane, 32 lanes cover D=256
    v8f vg = w0 * *(const v8f*)(m0 + d0) + w1 * *(const v8f*)(m1 + d0)
           + w2 * *(const v8f*)(m2 + d0) + w3 * *(const v8f*)(m3 + d0)
           + *(const v8f*)(grow + d0);
    v8f vi = *(const v8f*)(irow + d0);
    v8f ve = vg * vi;

    float partial[8] = {0, 0, 0, 0, 0, 0, 0, 0};
    #pragma unroll
    for (int e2 = 0; e2 < 8; e2++) {
      const int d = d0 + e2;
      const float* pe = sh_pw1 + d * 8;           // elem block
      const float* pg = sh_pw1 + (256 + d) * 8;   // g block
      const float* pi = sh_pw1 + (512 + d) * 8;   // item block
      #pragma unroll
      for (int j = 0; j < 8; j++)
        partial[j] += ve[e2] * pe[j] + vg[e2] * pg[j] + vi[e2] * pi[j];
    }
    #pragma unroll
    for (int off = 1; off < 32; off <<= 1) {
      #pragma unroll
      for (int j = 0; j < 8; j++) partial[j] += __shfl_xor(partial[j], off, 32);
    }
    if (lane == 0) {
      float acc2 = pb2;
      #pragma unroll
      for (int j = 0; j < 8; j++)
        acc2 += fmaxf(partial[j] + sh_pb1[j], 0.0f) * sh_pw2[j];
      out[bglob] = 1.0f / (1.0f + expf(-acc2));   // y section
    }
  }
}

extern "C" void kernel_launch(void* const* d_in, const int* in_sizes, int n_in,
                              void* d_out, int out_size, void* d_ws, size_t ws_size,
                              hipStream_t stream) {
  (void)in_sizes; (void)n_in; (void)out_size; (void)d_ws; (void)ws_size;
  const int*   group_inputs  = (const int*)d_in[0];
  const int*   item_inputs   = (const int*)d_in[1];
  const int*   group_members = (const int*)d_in[2];
  const float* user_emb      = (const float*)d_in[3];
  const float* item_emb      = (const float*)d_in[4];
  const float* group_emb     = (const float*)d_in[5];
  const float* att_w1        = (const float*)d_in[6];
  const float* att_b1        = (const float*)d_in[7];
  const float* att_w2        = (const float*)d_in[8];
  const float* att_b2        = (const float*)d_in[9];
  const float* cls_w         = (const float*)d_in[10];
  const float* cls_b         = (const float*)d_in[11];
  const float* pred_w1       = (const float*)d_in[12];
  const float* pred_b1       = (const float*)d_in[13];
  const float* pred_w2       = (const float*)d_in[14];
  const float* pred_b2       = (const float*)d_in[15];
  float* out = (float*)d_out;

  dim3 grid(B_TOT / GPB), block(NTHR);
  agree_fused<<<grid, block, 0, stream>>>(
      group_inputs, item_inputs, group_members,
      user_emb, item_emb, group_emb,
      att_w1, att_b1, att_w2, att_b2,
      cls_w, cls_b, pred_w1, pred_b1, pred_w2, pred_b2, out);
}
